// Mamba_17119739642026
// MI455X (gfx1250) — matrix operand order repaired and checked by reading, not compiled, not run on hardware
//
#include <hip/hip_runtime.h>
#include <math.h>

// ---- model dims (fixed by the reference) ----
#define NBLK   4
#define HDIM   256
#define DIM_I  512
#define DSTATE 16
#define DCONV  4
#define DRANK  16
#define INDIM  128
#define OUTDIM 10
#define NBATCH 4
#define SEQL   8192
#define MROWS  (NBATCH * SEQL)   // 32768
#define NDBC   (DRANK + 2 * DSTATE)  // 48

// ---- CDNA5 WMMA vector types (probe-verified signatures) ----
typedef __attribute__((ext_vector_type(16))) _Float16 f16x16;
typedef __attribute__((ext_vector_type(8)))  _Float16 f16x8;
typedef __attribute__((ext_vector_type(4)))  _Float16 f16x4;
typedef __attribute__((ext_vector_type(8)))  float    f32x8;

enum { ACT_NONE = 0, ACT_SOFTPLUS = 1, ACT_GELU = 2 };

// =====================================================================
// GEMM:  C[M,N] = act( A[M,K] @ W[N,K]^T + bias )
//   All shapes compile-time: guards fold away, K loop fully unrolled.
//   128 threads = 4 waves; block tile 128(M) x 64(N); K stepped by 32.
//   Per wave / K-step: 2 A-frags x 4 B-frags -> 8 v_wmma, f32 accumulate.
//   Ping-pong LDS (24 KB) so next tile's global loads overlap WMMA.
// =====================================================================
template <int ACT, bool HAS_BIAS, int K, int N, int LDA, int LDC>
__global__ void __launch_bounds__(128)
gemm_wmma(const float* __restrict__ A, const float* __restrict__ W,
          const float* __restrict__ bias, float* __restrict__ C)
{
    constexpr int KT = (K + 31) / 32;     // K tiles (zero-padded)
    __shared__ __align__(16) _Float16 As[2][128][32];   // [buf][m][k]
    __shared__ __align__(16) _Float16 Bs[2][64][32];    // [buf][n][k]

    const int tid   = threadIdx.x;
    const int wave  = tid >> 5;
    const int lane  = tid & 31;
    const int tileM = blockIdx.y * 128;
    const int tileN = blockIdx.x * 64;

    const f32x8 zero = {0.f, 0.f, 0.f, 0.f, 0.f, 0.f, 0.f, 0.f};
    f32x8 acc[2][4] = {{zero, zero, zero, zero}, {zero, zero, zero, zero}};

    // A: one 32-float row chunk per thread; B: half-row (16 floats) per thread
    const float* Arow_p = A + (size_t)(tileM + tid) * LDA;
    const int    bn_ld  = tid >> 1;
    const int    bk_ld  = (tid & 1) * 16;
    const int    gn     = tileN + bn_ld;
    const bool   nvald  = (N % 64 == 0) || (gn < N);   // folds unless N==48
    const float* Wrow_p = W + (size_t)gn * K;

    auto stage = [&](int t, int buf) {
        const int k0 = t * 32;
        #pragma unroll
        for (int j = 0; j < 32; j += 4) {
            float4 fa = {0.f, 0.f, 0.f, 0.f};
            if (k0 + j < K) fa = *(const float4*)(Arow_p + k0 + j);   // folds
            f16x4 ha = {(_Float16)fa.x, (_Float16)fa.y, (_Float16)fa.z, (_Float16)fa.w};
            *(f16x4*)&As[buf][tid][j] = ha;
        }
        #pragma unroll
        for (int j = 0; j < 16; j += 4) {
            float4 fw = {0.f, 0.f, 0.f, 0.f};
            if (nvald && (k0 + bk_ld + j < K))
                fw = *(const float4*)(Wrow_p + k0 + bk_ld + j);
            f16x4 hw = {(_Float16)fw.x, (_Float16)fw.y, (_Float16)fw.z, (_Float16)fw.w};
            *(f16x4*)&Bs[buf][bn_ld][bk_ld + j] = hw;
        }
    };

    union Frag { f16x16 v; f16x8 h[2]; };

    auto compute = [&](int buf) {
        // A frag (ISA 7.12.2): lanes 0-15 row m K[0..7]/K[16..23];
        //                      lanes 16-31 row m K[8..15]/K[24..31]
        const int arow = wave * 32 + (lane & 15);
        const int kh   = lane >> 4;
        Frag a0, a1;
        a0.h[0] = *(const f16x8*)&As[buf][arow][8 * kh];
        a0.h[1] = *(const f16x8*)&As[buf][arow][16 + 8 * kh];
        a1.h[0] = *(const f16x8*)&As[buf][arow + 16][8 * kh];
        a1.h[1] = *(const f16x8*)&As[buf][arow + 16][16 + 8 * kh];
        #pragma unroll
        for (int f = 0; f < 4; ++f) {
            // B frag: lanes 0-15 col n K[0..7]/K[16..23]; lanes 16-31 K[8..15]/K[24..31]
            Frag b;
            b.h[0] = *(const f16x8*)&Bs[buf][f * 16 + (lane & 15)][8 * kh];
            b.h[1] = *(const f16x8*)&Bs[buf][f * 16 + (lane & 15)][16 + 8 * kh];
            acc[0][f] = __builtin_amdgcn_wmma_f32_16x16x32_f16(
                false, a0.v, false, b.v, (short)0, acc[0][f], false, false);
            acc[1][f] = __builtin_amdgcn_wmma_f32_16x16x32_f16(
                false, a1.v, false, b.v, (short)0, acc[1][f], false, false);
        }
    };

    stage(0, 0);
    __syncthreads();
    #pragma unroll
    for (int t = 0; t < KT; ++t) {
        if (t + 2 < KT)                                    // folds at compile time
            __builtin_prefetch(Arow_p + (t + 2) * 32, 0, 1);  // global_prefetch_b8
        if (t + 1 < KT) stage(t + 1, (t + 1) & 1);
        compute(t & 1);
        __syncthreads();
    }

    // epilogue: C layout = VGPR r -> row r (+8 for lanes 16-31), col = lane&15
    #pragma unroll
    for (int af = 0; af < 2; ++af) {
        const int crow0 = tileM + wave * 32 + af * 16 + (lane >> 4) * 8;
        #pragma unroll
        for (int f = 0; f < 4; ++f) {
            const int col = tileN + f * 16 + (lane & 15);
            if ((N % 64 == 0) || col < N) {
                const float bv = HAS_BIAS ? bias[col] : 0.f;
                #pragma unroll
                for (int r = 0; r < 8; ++r) {
                    float v = acc[af][f][r] + bv;
                    if (ACT == ACT_SOFTPLUS)
                        v = (v > 20.f) ? v : log1pf(__expf(v));
                    else if (ACT == ACT_GELU)
                        v = 0.5f * v * (1.f + erff(v * 0.70710678118f));
                    C[(size_t)(crow0 + r) * LDC + col] = v;
                }
            }
        }
    }
}

// =====================================================================
// LayerNorm over last dim (256): one wave per row, 8 rows per block.
// =====================================================================
__global__ void __launch_bounds__(256)
layernorm_kernel(const float* __restrict__ x, const float* __restrict__ w,
                 const float* __restrict__ b, float* __restrict__ out)
{
    const int wave = threadIdx.x >> 5, lane = threadIdx.x & 31;
    const int row  = blockIdx.x * 8 + wave;
    const float* px = x + (size_t)row * HDIM;

    float4 v0 = *(const float4*)(px + lane * 8);
    float4 v1 = *(const float4*)(px + lane * 8 + 4);
    float s = v0.x + v0.y + v0.z + v0.w + v1.x + v1.y + v1.z + v1.w;
    #pragma unroll
    for (int off = 16; off > 0; off >>= 1) s += __shfl_xor(s, off, 32);
    const float mu = s * (1.f / HDIM);

    float d0x=v0.x-mu, d0y=v0.y-mu, d0z=v0.z-mu, d0w=v0.w-mu;
    float d1x=v1.x-mu, d1y=v1.y-mu, d1z=v1.z-mu, d1w=v1.w-mu;
    float ss = d0x*d0x + d0y*d0y + d0z*d0z + d0w*d0w
             + d1x*d1x + d1y*d1y + d1z*d1z + d1w*d1w;
    #pragma unroll
    for (int off = 16; off > 0; off >>= 1) ss += __shfl_xor(ss, off, 32);
    const float rstd = rsqrtf(ss * (1.f / HDIM) + 1e-5f);

    float4 w0 = *(const float4*)(w + lane * 8);
    float4 w1 = *(const float4*)(w + lane * 8 + 4);
    float4 b0 = *(const float4*)(b + lane * 8);
    float4 b1 = *(const float4*)(b + lane * 8 + 4);
    float4 o0, o1;
    o0.x = d0x*rstd*w0.x + b0.x;  o0.y = d0y*rstd*w0.y + b0.y;
    o0.z = d0z*rstd*w0.z + b0.z;  o0.w = d0w*rstd*w0.w + b0.w;
    o1.x = d1x*rstd*w1.x + b1.x;  o1.y = d1y*rstd*w1.y + b1.y;
    o1.z = d1z*rstd*w1.z + b1.z;  o1.w = d1w*rstd*w1.w + b1.w;
    float* po = out + (size_t)row * HDIM + lane * 8;
    *(float4*)(po)     = o0;
    *(float4*)(po + 4) = o1;
}

// =====================================================================
// Causal depthwise conv (DC=4) over the xc half of xz + SiLU; compacts
// xc from stride-1024 rows into a contiguous [M, DI] buffer.
// =====================================================================
__global__ void __launch_bounds__(256)
conv_silu_kernel(const float* __restrict__ xz, const float* __restrict__ cw,
                 const float* __restrict__ cb, float* __restrict__ xconv)
{
    const int idx = blockIdx.x * 256 + threadIdx.x;   // over M*DI
    const int d   = idx & (DIM_I - 1);
    const int bl  = idx >> 9;
    const int l   = bl & (SEQL - 1);

    float acc = cb[d];
    #pragma unroll
    for (int j = 0; j < DCONV; ++j) {
        const int ls = l - (DCONV - 1) + j;
        if (ls >= 0)
            acc += cw[d * DCONV + j] * xz[(size_t)(bl - (DCONV - 1) + j) * (2 * DIM_I) + d];
    }
    const float sig = 1.f / (1.f + __expf(-acc));
    xconv[idx] = acc * sig;
}

// =====================================================================
// Selective scan: sequential over L; one thread per (b, d) channel,
// 16-deep SSM state in registers; B_t/C_t (32 shared floats) staged in LDS.
// Fuses D-skip and the SiLU(z) gate; writes y [M, DI].
// =====================================================================
__global__ void __launch_bounds__(128)
scan_kernel(const float* __restrict__ dt, const float* __restrict__ xconv,
            const float* __restrict__ dbc, const float* __restrict__ xz,
            const float* __restrict__ A_log, const float* __restrict__ D_skip,
            float* __restrict__ y)
{
    const int b = blockIdx.y;
    const int d = blockIdx.x * 128 + threadIdx.x;
    __shared__ float sBC[2 * DSTATE];

    float Arow[DSTATE];
    #pragma unroll
    for (int s = 0; s < DSTATE; ++s) Arow[s] = -__expf(A_log[d * DSTATE + s]);
    const float Dd = D_skip[d];

    float st[DSTATE];
    #pragma unroll
    for (int s = 0; s < DSTATE; ++s) st[s] = 0.f;

    for (int l = 0; l < SEQL; ++l) {
        const size_t bl = (size_t)b * SEQL + l;
        if (threadIdx.x < 2 * DSTATE)
            sBC[threadIdx.x] = dbc[bl * NDBC + DRANK + threadIdx.x];
        __syncthreads();

        const float dtv = dt[bl * DIM_I + d];
        const float xv  = xconv[bl * DIM_I + d];
        const float zv  = xz[bl * (2 * DIM_I) + DIM_I + d];
        const float dtx = dtv * xv;

        float acc = 0.f;
        #pragma unroll
        for (int s = 0; s < DSTATE; ++s) {
            const float dA = __expf(dtv * Arow[s]);     // v_exp_f32 (TRANS)
            st[s] = st[s] * dA + dtx * sBC[s];
            acc  += st[s] * sBC[DSTATE + s];
        }
        const float yv  = acc + xv * Dd;
        const float sig = 1.f / (1.f + __expf(-zv));
        y[bl * DIM_I + d] = yv * (zv * sig);
        __syncthreads();
    }
}

// =====================================================================
// GLU gate + residual:  h = g[:, :H] * sigmoid(g[:, H:]) + h
// =====================================================================
__global__ void __launch_bounds__(256)
glu_residual_kernel(const float* __restrict__ g, float* __restrict__ h)
{
    const size_t row = blockIdx.x;
    const int    col = threadIdx.x;
    const float a  = g[row * (2 * HDIM) + col];
    const float gt = g[row * (2 * HDIM) + HDIM + col];
    const float sig = 1.f / (1.f + __expf(-gt));
    h[row * HDIM + col] = a * sig + h[row * HDIM + col];
}

// =====================================================================
// Mean-pool over L, then decode + softmax.
// =====================================================================
__global__ void __launch_bounds__(128)
pool_kernel(const float* __restrict__ h, float* __restrict__ pooled)
{
    const int b   = blockIdx.y;
    const int col = blockIdx.x * 128 + threadIdx.x;
    float s = 0.f;
    for (int l = 0; l < SEQL; ++l)
        s += h[((size_t)b * SEQL + l) * HDIM + col];
    pooled[b * HDIM + col] = s * (1.f / SEQL);
}

__global__ void __launch_bounds__(64)
decode_kernel(const float* __restrict__ pooled, const float* __restrict__ dw,
              const float* __restrict__ db, float* __restrict__ out)
{
    __shared__ float lg[NBATCH][OUTDIM];
    const int t = threadIdx.x;
    if (t < NBATCH * OUTDIM) {
        const int b = t / OUTDIM, o = t % OUTDIM;
        float s = db[o];
        for (int k = 0; k < HDIM; ++k)
            s += pooled[b * HDIM + k] * dw[o * HDIM + k];
        lg[b][o] = s;
    }
    __syncthreads();
    if (t < NBATCH) {
        float mx = -1e30f;
        for (int o = 0; o < OUTDIM; ++o) mx = fmaxf(mx, lg[t][o]);
        float e[OUTDIM], sum = 0.f;
        for (int o = 0; o < OUTDIM; ++o) { e[o] = __expf(lg[t][o] - mx); sum += e[o]; }
        const float inv = 1.f / sum;
        for (int o = 0; o < OUTDIM; ++o) out[t * OUTDIM + o] = e[o] * inv;
    }
}

// =====================================================================
extern "C" void kernel_launch(void* const* d_in, const int* in_sizes, int n_in,
                              void* d_out, int out_size, void* d_ws, size_t ws_size,
                              hipStream_t stream)
{
    (void)in_sizes; (void)n_in; (void)out_size; (void)ws_size;

    const float* x         = (const float*)d_in[0];
    const float* enc_w     = (const float*)d_in[1];
    const float* enc_b     = (const float*)d_in[2];
    const float* norm_w    = (const float*)d_in[3];
    const float* norm_b    = (const float*)d_in[4];
    const float* in_proj_w = (const float*)d_in[5];
    const float* conv_w    = (const float*)d_in[6];
    const float* conv_b    = (const float*)d_in[7];
    const float* xproj_w   = (const float*)d_in[8];
    const float* dtproj_w  = (const float*)d_in[9];
    const float* dtproj_b  = (const float*)d_in[10];
    const float* A_log     = (const float*)d_in[11];
    const float* D_skip    = (const float*)d_in[12];
    const float* outproj_w = (const float*)d_in[13];
    const float* glu_w     = (const float*)d_in[14];
    const float* glu_b     = (const float*)d_in[15];
    const float* dec_w     = (const float*)d_in[16];
    const float* dec_b     = (const float*)d_in[17];

    // workspace layout (f32), ~410 MB total; g reuses xz, gelu-out reuses v
    float* ws = (float*)d_ws;
    size_t off = 0;
    float* hbuf   = ws + off; off += (size_t)MROWS * HDIM;          // residual stream
    float* vbuf   = ws + off; off += (size_t)MROWS * HDIM;          // ln out / gelu out
    float* xzbuf  = ws + off; off += (size_t)MROWS * 2 * DIM_I;     // xz / glu pre-act
    float* xconv  = ws + off; off += (size_t)MROWS * DIM_I;
    float* dbcbuf = ws + off; off += (size_t)MROWS * NDBC;
    float* dtbuf  = ws + off; off += (size_t)MROWS * DIM_I;
    float* ybuf   = ws + off; off += (size_t)MROWS * DIM_I;
    float* pooled = ws + off; off += (size_t)NBATCH * HDIM;

    const dim3 blk(128);
    const int mt = MROWS / 128;   // 256 M-tiles

    // encoder: h = x @ enc_w^T + enc_b
    gemm_wmma<ACT_NONE, true, INDIM, HDIM, INDIM, HDIM>
        <<<dim3(HDIM / 64, mt), blk, 0, stream>>>(x, enc_w, enc_b, hbuf);

    for (int i = 0; i < NBLK; ++i) {
        layernorm_kernel<<<MROWS / 8, 256, 0, stream>>>(
            hbuf, norm_w + i * HDIM, norm_b + i * HDIM, vbuf);

        // xz = v @ in_proj_w^T  [M, 1024]
        gemm_wmma<ACT_NONE, false, HDIM, 2 * DIM_I, HDIM, 2 * DIM_I>
            <<<dim3(2 * DIM_I / 64, mt), blk, 0, stream>>>(
            vbuf, in_proj_w + (size_t)i * 2 * DIM_I * HDIM, nullptr, xzbuf);

        conv_silu_kernel<<<(MROWS * DIM_I) / 256, 256, 0, stream>>>(
            xzbuf, conv_w + i * DIM_I * DCONV, conv_b + i * DIM_I, xconv);

        // dbc = xconv @ xproj_w^T  [M, 48]
        gemm_wmma<ACT_NONE, false, DIM_I, NDBC, DIM_I, NDBC>
            <<<dim3(1, mt), blk, 0, stream>>>(
            xconv, xproj_w + (size_t)i * NDBC * DIM_I, nullptr, dbcbuf);

        // dt = softplus(dbc[:, :16] @ dtproj_w^T + dtproj_b)  [M, 512]
        gemm_wmma<ACT_SOFTPLUS, true, DRANK, DIM_I, NDBC, DIM_I>
            <<<dim3(DIM_I / 64, mt), blk, 0, stream>>>(
            dbcbuf, dtproj_w + (size_t)i * DIM_I * DRANK, dtproj_b + i * DIM_I, dtbuf);

        scan_kernel<<<dim3(DIM_I / 128, NBATCH), 128, 0, stream>>>(
            dtbuf, xconv, dbcbuf, xzbuf,
            A_log + (size_t)i * DIM_I * DSTATE, D_skip + i * DIM_I, ybuf);

        // gelu(y @ outproj_w^T)  [M, 256]
        gemm_wmma<ACT_GELU, false, DIM_I, HDIM, DIM_I, HDIM>
            <<<dim3(HDIM / 64, mt), blk, 0, stream>>>(
            ybuf, outproj_w + (size_t)i * HDIM * DIM_I, nullptr, vbuf);

        // g = v @ glu_w^T + glu_b  [M, 512]  (reuses xz buffer)
        gemm_wmma<ACT_NONE, true, HDIM, 2 * HDIM, HDIM, 2 * HDIM>
            <<<dim3(2 * HDIM / 64, mt), blk, 0, stream>>>(
            vbuf, glu_w + (size_t)i * 2 * HDIM * HDIM, glu_b + i * 2 * HDIM, xzbuf);

        glu_residual_kernel<<<MROWS, 256, 0, stream>>>(xzbuf, hbuf);
    }

    pool_kernel<<<dim3(HDIM / 128, NBATCH), 128, 0, stream>>>(hbuf, pooled);
    decode_kernel<<<1, 64, 0, stream>>>(pooled, dec_w, dec_b, (float*)d_out);
}
